// Transformer_13383118094606
// MI455X (gfx1250) — compile-verified, matrix operand order checked by reading, not run on hardware
//
#include <hip/hip_runtime.h>
#include <hip/hip_bf16.h>
#include <stdint.h>

typedef __attribute__((ext_vector_type(16))) __bf16 v16bf;
typedef __attribute__((ext_vector_type(8)))  float  v8f;
typedef __attribute__((ext_vector_type(4)))  unsigned int v4u;
typedef __attribute__((ext_vector_type(8)))  int    v8i;
typedef __attribute__((ext_vector_type(4)))  int    v4i;

#define S_LEN   2048
#define BATCH   2
#define TOKENS  (S_LEN * BATCH)
#define HID     512
#define NH      8
#define QKD     96      // UP_HEAD + ROPE_HEAD
#define VD      64      // UP_HEAD

// ---------------------------------------------------------------------------
// Tensor Data Mover: 2D tile (f32) global -> LDS.
// D# per CDNA5 ISA ch.8: group0 = {flags, lds_addr, global_addr, type=2},
// group1 = {data_size=4B, tensor_dim0/1 (OOB clamp), tile_dim0/1, dim0 stride}.
// Issued per-wave (EXEC ignored); tracked by TENSORcnt.
// ---------------------------------------------------------------------------
__device__ __forceinline__ void tdm_load_2d(unsigned lds_addr, const float* gaddr,
                                            int tensor_w, int tile_w, int tile_h,
                                            int stride_elems)
{
    const unsigned long long ga = (unsigned long long)(uintptr_t)gaddr;
    v4u g0;
    g0[0] = 1u;                                               // count=1, user desc
    g0[1] = lds_addr;                                         // LDS byte offset
    g0[2] = (unsigned)(ga & 0xFFFFFFFFu);                     // global_addr[31:0]
    g0[3] = (unsigned)((ga >> 32) & 0x1FFFFFFu) | (2u << 30); // addr[56:32] | type=2
    v8i g1;
    g1[0] = (int)(2u << 16);                                  // data_size = 4 bytes
    g1[1] = (int)(((unsigned)tensor_w & 0xFFFFu) << 16);      // tensor_dim0[15:0]
    g1[2] = (int)((((unsigned)tensor_w >> 16) & 0xFFFFu) |
                  (((unsigned)tile_h & 0xFFFFu) << 16));      // dim0 hi | tensor_dim1 lo
    g1[3] = (int)((((unsigned)tile_h >> 16) & 0xFFFFu) |
                  (((unsigned)tile_w & 0xFFFFu) << 16));      // dim1 hi | tile_dim0
    g1[4] = (int)((unsigned)tile_h & 0xFFFFu);                // tile_dim1 (tile_dim2=0)
    g1[5] = (int)(unsigned)stride_elems;                      // tensor_dim0_stride[31:0]
    g1[6] = 0;                                                // stride hi | dim1_stride lo
    g1[7] = 0;
    const v4i gz = {0, 0, 0, 0};
#if defined(__clang_major__) && (__clang_major__ >= 23)
    const v8i gz8 = {0, 0, 0, 0, 0, 0, 0, 0};
    __builtin_amdgcn_tensor_load_to_lds(g0, g1, gz, gz, gz8, 0);
#else
    __builtin_amdgcn_tensor_load_to_lds(g0, g1, gz, gz, 0);
#endif
}

// ---------------------------------------------------------------------------
// bf16 WMMA GEMM with TDM-staged LDS tiles: C[M,N] = A[M,K] @ B[K,N] (f32 mem)
// Block (32,4): 64x64 output tile; wave ty owns a 16x64 strip (4 WMMA / K-step).
// K must be a multiple of 32; M a multiple of 64; N arbitrary (TDM zero-pads).
// mode 0: C = acc            mode 1: C = acc + resid
// mode 2: C += rowscale[row*rs_stride] * acc
// mode 3: C = sigmoid(acc)   mode 4: C += acc
// ---------------------------------------------------------------------------
__global__ void gemm_wmma_kernel(const float* __restrict__ A,
                                 const float* __restrict__ Bm,
                                 float* __restrict__ C,
                                 int M, int N, int K,
                                 const float* __restrict__ resid,
                                 const float* __restrict__ rowscale, int rs_stride,
                                 int mode)
{
    __shared__ float sA[64 * 32];   // 64 rows x 32 k   (8 KB)
    __shared__ float sB[32 * 64];   // 32 k   x 64 cols (8 KB)

    const int lane = threadIdx.x;
    const int ty   = threadIdx.y;
    const int row0 = blockIdx.y * 64;
    const int col0 = blockIdx.x * 64;

    const int mr = lane & 15;
    const int ko = (lane < 16) ? 0 : 8;
    const int mo = (lane < 16) ? 0 : 8;

    const unsigned ldsA = (unsigned)(uintptr_t)sA;  // low 32 bits = LDS byte offset
    const unsigned ldsB = (unsigned)(uintptr_t)sB;
    const int bw = (N - col0 < 64) ? (N - col0) : 64;   // valid B width (OOB -> 0)

    v8f acc[4] = {};

    for (int k0 = 0; k0 < K; k0 += 32) {
        if (ty == 0) {
            tdm_load_2d(ldsA, A + (size_t)row0 * K + k0, 32, 32, 64, K);
            tdm_load_2d(ldsB, Bm + (size_t)k0 * N + col0, bw, 64, 32, N);
            __builtin_amdgcn_s_wait_tensorcnt(0);
        }
        __syncthreads();

        // A fragment: rows ty*16 + (lane&15), 16x32 bf16
        v16bf af;
        {
            const float* ar = &sA[(ty * 16 + mr) * 32];
#pragma unroll
            for (int i = 0; i < 8; ++i) {
                af[i]     = (__bf16)ar[ko + i];
                af[i + 8] = (__bf16)ar[16 + ko + i];
            }
        }
        // 4 B fragments (32x16 each) and WMMAs
#pragma unroll
        for (int c = 0; c < 4; ++c) {
            v16bf bf;
            const float* br = &sB[c * 16 + mr];
#pragma unroll
            for (int i = 0; i < 8; ++i) {
                bf[i]     = (__bf16)br[(ko + i) * 64];
                bf[i + 8] = (__bf16)br[(16 + ko + i) * 64];
            }
            acc[c] = __builtin_amdgcn_wmma_f32_16x16x32_bf16(false, af, false, bf,
                                                             (short)0, acc[c], false, false);
        }
        __syncthreads();   // before TDM overwrites tiles
    }

#pragma unroll
    for (int c = 0; c < 4; ++c) {
        const int cn = col0 + c * 16 + mr;
        if (cn < N) {
#pragma unroll
            for (int r = 0; r < 8; ++r) {
                const int row = row0 + ty * 16 + mo + r;
                const size_t idx = (size_t)row * N + cn;
                const float v = acc[c][r];
                if (mode == 0)      C[idx] = v;
                else if (mode == 1) C[idx] = v + resid[idx];
                else if (mode == 2) C[idx] += rowscale[(size_t)row * rs_stride] * v;
                else if (mode == 3) C[idx] = 1.0f / (1.0f + __expf(-v));
                else                C[idx] += v;
            }
        }
    }
}

// ---------------------------------------------------------------------------
// RMSNorm: one block per token
// ---------------------------------------------------------------------------
__global__ void rmsnorm_kernel(const float* __restrict__ in, int istride,
                               const float* __restrict__ w,
                               float* __restrict__ out, int ostride, int D)
{
    const int t = blockIdx.x;
    const float* x = in + (size_t)t * istride;
    float* o = out + (size_t)t * ostride;
    __shared__ float sbuf[256];
    float ss = 0.0f;
    for (int i = threadIdx.x; i < D; i += blockDim.x) { float v = x[i]; ss += v * v; }
    sbuf[threadIdx.x] = ss;
    __syncthreads();
    for (int off = blockDim.x >> 1; off > 0; off >>= 1) {
        if ((int)threadIdx.x < off) sbuf[threadIdx.x] += sbuf[threadIdx.x + off];
        __syncthreads();
    }
    const float inv = rsqrtf(sbuf[0] / (float)D + 1.1920929e-07f);
    for (int i = threadIdx.x; i < D; i += blockDim.x) o[i] = x[i] * inv * w[i];
}

// ---------------------------------------------------------------------------
// RoPE (half-rotated, D=32): q applied in-place per head; k writes kr buffer
// ---------------------------------------------------------------------------
__global__ void rope_q_kernel(float* __restrict__ qcr)
{
    const int t = blockIdx.x;
    const int pos = t % S_LEN;
    const int h = threadIdx.x >> 4, i = threadIdx.x & 15;   // blockDim = 128
    const float invf = __expf(-(float)i * (9.210340371976184f / 16.0f));
    const float ang = (float)pos * invf;
    const float c = __cosf(ang), s = __sinf(ang);
    float* p = qcr + (size_t)t * (NH * QKD) + h * QKD + VD;
    const float x1 = p[i], x2 = p[i + 16];
    p[i]      = x1 * c - x2 * s;
    p[i + 16] = x2 * c + x1 * s;
}

__global__ void rope_k_kernel(const float* __restrict__ ckv, float* __restrict__ kr)
{
    const int t = blockIdx.x;
    const int pos = t % S_LEN;
    const int i = threadIdx.x;                               // blockDim = 16
    const float invf = __expf(-(float)i * (9.210340371976184f / 16.0f));
    const float ang = (float)pos * invf;
    const float c = __cosf(ang), s = __sinf(ang);
    const float* p = ckv + (size_t)t * 160 + 128;
    float* o = kr + (size_t)t * 32;
    const float x1 = p[i], x2 = p[i + 16];
    o[i]      = x1 * c - x2 * s;
    o[i + 16] = x2 * c + x1 * s;
}

// ---------------------------------------------------------------------------
// MLA attention, flash style. One wave per 16-query tile, KV chunks of 32.
// K = [k_c (64, from kvbuf) | rope(k_r) (32, from krbuf, shared over heads)]
// grid = (S/64, B*NH), block = (32,4)
// ---------------------------------------------------------------------------
__global__ void mla_attn_kernel(const float* __restrict__ qcr,
                                const float* __restrict__ kvbuf,
                                const float* __restrict__ krbuf,
                                float* __restrict__ obuf)
{
    const int lane = threadIdx.x;
    const int wv = threadIdx.y;
    const int q0 = (blockIdx.x * 4 + wv) * 16;
    const int b = blockIdx.y >> 3, h = blockIdx.y & 7;
    const size_t tq = (size_t)b * S_LEN;

    __shared__ float sS[4][16][33];
    __shared__ float sC[4][16];
    __shared__ float sL[4][16];

    const int mr = lane & 15;
    const int ko = (lane < 16) ? 0 : 8;
    const int mo = (lane < 16) ? 0 : 8;

    // Q fragments (16 x 96), rope already applied in qcr
    v16bf qf[3];
    {
        const float* qrow = qcr + (tq + q0 + mr) * (NH * QKD) + h * QKD;
#pragma unroll
        for (int j = 0; j < 3; ++j) {
            const float* p = qrow + j * 32 + ko;
#pragma unroll
            for (int i = 0; i < 8; ++i) {
                qf[j][i]     = (__bf16)p[i];
                qf[j][i + 8] = (__bf16)p[16 + i];
            }
        }
    }

    v8f oacc[4] = {};
    float m_run = -3.0e38f, l_run = 0.0f;
    const float scale = 0.10206207261596575f;   // 1/sqrt(96)

    for (int kv0 = 0; kv0 < S_LEN; kv0 += 32) {
        // ---- S tile = Q @ K^T (16 x 32), two 16-col halves ----
        v8f sacc[2] = {};
#pragma unroll
        for (int c = 0; c < 2; ++c) {
            const size_t tk = tq + kv0 + c * 16 + mr;        // kv row = column n
            const float* kc = kvbuf + tk * (NH * 2 * VD) + h * (2 * VD);
            const float* kr = krbuf + tk * 32;
#pragma unroll
            for (int j = 0; j < 3; ++j) {
                v16bf kf;
#pragma unroll
                for (int i = 0; i < 8; ++i) {
                    const int d0 = j * 32 + ko + i;
                    const int d1 = j * 32 + 16 + ko + i;
                    kf[i]     = (__bf16)((d0 < VD) ? kc[d0] : kr[d0 - VD]);
                    kf[i + 8] = (__bf16)((d1 < VD) ? kc[d1] : kr[d1 - VD]);
                }
                sacc[c] = __builtin_amdgcn_wmma_f32_16x16x32_bf16(false, qf[j], false, kf,
                                                                  (short)0, sacc[c], false, false);
            }
        }
        // ---- logits -> LDS ----
#pragma unroll
        for (int c = 0; c < 2; ++c)
#pragma unroll
            for (int r = 0; r < 8; ++r)
                sS[wv][mo + r][c * 16 + mr] = sacc[c][r] * scale;
        __syncthreads();
        // ---- online softmax row stats (lanes 0..15 own one row each) ----
        if (lane < 16) {
            float mx = m_run;
            for (int j = 0; j < 32; ++j) mx = fmaxf(mx, sS[wv][lane][j]);
            const float corr = __expf(m_run - mx);
            float ps = 0.0f;
            for (int j = 0; j < 32; ++j) {
                const float p = __expf(sS[wv][lane][j] - mx);
                sS[wv][lane][j] = p;
                ps += p;
            }
            l_run = l_run * corr + ps;
            m_run = mx;
            sC[wv][lane] = corr;
        }
        __syncthreads();
        // ---- rescale O accumulators ----
#pragma unroll
        for (int r = 0; r < 8; ++r) {
            const float cf = sC[wv][mo + r];
#pragma unroll
            for (int c = 0; c < 4; ++c) oacc[c][r] *= cf;
        }
        // ---- P fragment (16 x 32 bf16 A-matrix) from LDS ----
        v16bf pf;
#pragma unroll
        for (int i = 0; i < 8; ++i) {
            pf[i]     = (__bf16)sS[wv][mr][ko + i];
            pf[i + 8] = (__bf16)sS[wv][mr][16 + ko + i];
        }
        // ---- O += P @ V (32 x 64) ----
#pragma unroll
        for (int c = 0; c < 4; ++c) {
            v16bf vf;
#pragma unroll
            for (int i = 0; i < 8; ++i) {
                const size_t tv0 = tq + kv0 + ko + i;
                const size_t tv1 = tq + kv0 + 16 + ko + i;
                vf[i]     = (__bf16)kvbuf[tv0 * (NH * 2 * VD) + h * (2 * VD) + VD + c * 16 + mr];
                vf[i + 8] = (__bf16)kvbuf[tv1 * (NH * 2 * VD) + h * (2 * VD) + VD + c * 16 + mr];
            }
            oacc[c] = __builtin_amdgcn_wmma_f32_16x16x32_bf16(false, pf, false, vf,
                                                              (short)0, oacc[c], false, false);
        }
        __syncthreads();   // before next chunk overwrites sS
    }

    if (lane < 16) sL[wv][lane] = 1.0f / l_run;
    __syncthreads();
#pragma unroll
    for (int r = 0; r < 8; ++r) {
        const float inv = sL[wv][mo + r];
        const size_t trow = (tq + q0 + mo + r) * (NH * VD) + h * VD;
#pragma unroll
        for (int c = 0; c < 4; ++c)
            obuf[trow + c * 16 + mr] = oacc[c][r] * inv;
    }
}

// ---------------------------------------------------------------------------
// Small elementwise kernels
// ---------------------------------------------------------------------------
__global__ void topk_kernel(const float* __restrict__ scores, float* __restrict__ combine, int T)
{
    const int t = blockIdx.x * blockDim.x + threadIdx.x;
    if (t >= T) return;
    float s[16];
    bool used[16];
#pragma unroll
    for (int i = 0; i < 16; ++i) { s[i] = scores[t * 16 + i]; used[i] = false; }
    int   idx[4];
    float w[4], sum = 0.0f;
#pragma unroll
    for (int k = 0; k < 4; ++k) {
        int best = 0; float bv = -3.0e38f;
        for (int i = 0; i < 16; ++i)
            if (!used[i] && s[i] > bv) { bv = s[i]; best = i; }
        used[best] = true; idx[k] = best; w[k] = bv; sum += bv;
    }
#pragma unroll
    for (int i = 0; i < 16; ++i) combine[t * 16 + i] = 0.0f;
#pragma unroll
    for (int k = 0; k < 4; ++k) combine[t * 16 + idx[k]] = w[k] / sum;
}

__global__ void silu_mul_kernel(const float* __restrict__ g, const float* __restrict__ u,
                                float* __restrict__ h, size_t n)
{
    const size_t i = (size_t)blockIdx.x * blockDim.x + threadIdx.x;
    if (i >= n) return;
    const float gv = g[i];
    h[i] = (gv / (1.0f + __expf(-gv))) * u[i];
}

__global__ void zero_kernel(float* __restrict__ p, size_t n)
{
    const size_t i = (size_t)blockIdx.x * blockDim.x + threadIdx.x;
    if (i < n) p[i] = 0.0f;
}

__global__ void add2_kernel(const float* __restrict__ a, const float* __restrict__ b,
                            float* __restrict__ o, size_t n)
{
    const size_t i = (size_t)blockIdx.x * blockDim.x + threadIdx.x;
    if (i < n) o[i] = a[i] + b[i];
}

// ---------------------------------------------------------------------------
extern "C" void kernel_launch(void* const* d_in, const int* in_sizes, int n_in,
                              void* d_out, int out_size, void* d_ws, size_t ws_size,
                              hipStream_t stream)
{
    const int T = TOKENS;
    const float* hidden   = (const float*)d_in[0];
    const float* w_innorm = (const float*)d_in[1];
    const float* w_dq     = (const float*)d_in[2];
    const float* w_qnorm  = (const float*)d_in[3];
    const float* w_uq     = (const float*)d_in[4];
    const float* w_dkv    = (const float*)d_in[5];
    const float* w_kvnorm = (const float*)d_in[6];
    const float* w_ukv    = (const float*)d_in[7];
    const float* w_o      = (const float*)d_in[8];
    const float* w_pnorm  = (const float*)d_in[9];
    const float* w_gate   = (const float*)d_in[10];
    const float* sh_wg    = (const float*)d_in[11];
    const float* sh_wu    = (const float*)d_in[12];
    const float* sh_wd    = (const float*)d_in[13];
    const float* r_wg     = (const float*)d_in[14];
    const float* r_wu     = (const float*)d_in[15];
    const float* r_wd     = (const float*)d_in[16];
    float* out = (float*)d_out;

    // ---- workspace layout (floats), with dead-buffer aliasing ----
    float* ws      = (float*)d_ws;
    float* xn      = ws;                        // T*512 normalized input
    float* cq      = xn   + (size_t)T * 512;    // T*256
    float* cqn     = cq   + (size_t)T * 256;    // T*256
    float* qcr     = cqn  + (size_t)T * 256;    // T*768
    float* ckv     = qcr  + (size_t)T * 768;    // T*160
    float* ckvn    = ckv  + (size_t)T * 160;    // T*128
    float* kvb     = ckvn + (size_t)T * 128;    // T*1024
    float* kr      = kvb  + (size_t)T * 1024;   // T*32
    float* scores  = kr   + (size_t)T * 32;     // T*16
    float* combine = scores + (size_t)T * 16;   // T*16
    float* hbuf    = combine + (size_t)T * 16;  // T*256
    float* xres    = hbuf + (size_t)T * 256;    // T*512
    // aliases of buffers dead by the time these are used:
    float* obuf   = xn;    // attention output (T*512)
    float* gbuf   = cq;    // MLP gate       (T*256)
    float* ubuf   = cqn;   // MLP up         (T*256)
    float* x2     = kvb;   // post-norm      (T*512)
    float* routed = qcr;   // MoE accum      (T*512, fits in T*768)

    const dim3 gblk(32, 4);
    auto gemm = [&](const float* A, const float* Bw, float* C, int M, int N, int K,
                    const float* resid, const float* rowscale, int rs, int mode) {
        dim3 grid((N + 63) / 64, (M + 63) / 64);
        gemm_wmma_kernel<<<grid, gblk, 0, stream>>>(A, Bw, C, M, N, K, resid, rowscale, rs, mode);
    };

    // ---- MLA front end ----
    rmsnorm_kernel<<<T, 256, 0, stream>>>(hidden, 512, w_innorm, xn, 512, 512);
    gemm(xn,   w_dq,  cq,  T, 256,  512, nullptr, nullptr, 0, 0);
    rmsnorm_kernel<<<T, 256, 0, stream>>>(cq, 256, w_qnorm, cqn, 256, 256);
    gemm(cqn,  w_uq,  qcr, T, 768,  256, nullptr, nullptr, 0, 0);
    gemm(xn,   w_dkv, ckv, T, 160,  512, nullptr, nullptr, 0, 0);
    rmsnorm_kernel<<<T, 256, 0, stream>>>(ckv, 160, w_kvnorm, ckvn, 128, 128);
    gemm(ckvn, w_ukv, kvb, T, 1024, 128, nullptr, nullptr, 0, 0);
    rope_q_kernel<<<T, 128, 0, stream>>>(qcr);
    rope_k_kernel<<<T, 16, 0, stream>>>(ckv, kr);

    // ---- attention ----
    mla_attn_kernel<<<dim3(S_LEN / 64, BATCH * NH), gblk, 0, stream>>>(qcr, kvb, kr, obuf);

    // ---- output proj + residual, post-norm ----
    gemm(obuf, w_o, xres, T, 512, 512, hidden, nullptr, 0, 1);
    rmsnorm_kernel<<<T, 256, 0, stream>>>(xres, 512, w_pnorm, x2, 512, 512);

    // ---- gating ----
    gemm(x2, w_gate, scores, T, 16, 512, nullptr, nullptr, 0, 3);   // sigmoid
    topk_kernel<<<(T + 255) / 256, 256, 0, stream>>>(scores, combine, T);

    // ---- MoE: shared expert accumulated into routed, then 16 routed experts ----
    const size_t n512 = (size_t)T * 512, n256 = (size_t)T * 256;
    zero_kernel<<<(unsigned)((n512 + 255) / 256), 256, 0, stream>>>(routed, n512);

    gemm(x2, sh_wg, gbuf, T, 256, 512, nullptr, nullptr, 0, 0);
    gemm(x2, sh_wu, ubuf, T, 256, 512, nullptr, nullptr, 0, 0);
    silu_mul_kernel<<<(unsigned)((n256 + 255) / 256), 256, 0, stream>>>(gbuf, ubuf, hbuf, n256);
    gemm(hbuf, sh_wd, routed, T, 512, 256, nullptr, nullptr, 0, 4);

    for (int e = 0; e < 16; ++e) {
        const float* wg = r_wg + (size_t)e * 512 * 256;
        const float* wu = r_wu + (size_t)e * 512 * 256;
        const float* wd = r_wd + (size_t)e * 256 * 512;
        gemm(x2, wg, gbuf, T, 256, 512, nullptr, nullptr, 0, 0);
        gemm(x2, wu, ubuf, T, 256, 512, nullptr, nullptr, 0, 0);
        silu_mul_kernel<<<(unsigned)((n256 + 255) / 256), 256, 0, stream>>>(gbuf, ubuf, hbuf, n256);
        gemm(hbuf, wd, routed, T, 512, 256, nullptr, combine + e, 16, 2);
    }

    // ---- final: out = routed(+shared) + residual2 ----
    add2_kernel<<<(unsigned)((n512 + 255) / 256), 256, 0, stream>>>(routed, xres, out, n512);
}
